// MultiHeadAttention_23467701305746
// MI455X (gfx1250) — matrix operand order repaired and checked by reading, not compiled, not run on hardware
//
#include <hip/hip_runtime.h>
#include <hip/hip_bf16.h>

// ---------------------------------------------------------------------------
// Types for CDNA5 WMMA / TDM
// ---------------------------------------------------------------------------
typedef __attribute__((ext_vector_type(16))) __bf16 v16bf;
typedef __attribute__((ext_vector_type(8)))  float  v8f;
typedef __attribute__((ext_vector_type(4)))  unsigned int v4u;
typedef __attribute__((ext_vector_type(8)))  int    v8i;
typedef __attribute__((ext_vector_type(4)))  int    v4i;

union FragBF {
    unsigned int u[8];
    v16bf        v;
};

__device__ __forceinline__ unsigned short f2bf(float f) {
    union { float f; unsigned int u; } c;
    c.f = f;
    unsigned int r = c.u + 0x7FFFu + ((c.u >> 16) & 1u);  // round to nearest even
    return (unsigned short)(r >> 16);
}

// A-matrix (16x32 bf16) K index for VGPR v, lane-half h (per ISA 7.12.2)
__device__ __forceinline__ int kA(int v, int half) {
    return (v < 4) ? (half * 8 + 2 * v) : (16 + half * 8 + 2 * (v - 4));
}

#define N_HIDDEN 512
#define SEQ      1024
#define BATCH    8
#define HEADS    16
#define HDIM     32
#define ROWS     (BATCH * SEQ)          // 8192
#define X_ELEMS  (ROWS * N_HIDDEN)      // 4194304
#define W_ELEMS  (N_HIDDEN * N_HIDDEN)  // 262144

// ---------------------------------------------------------------------------
// TDM: issue a 2D tile load (global -> LDS).  D# layout per CDNA5 ISA ch.8.
//   w0 packs: data_size=2B, pad_enable, pad_interval (row-sized), pad_amount=1DW
// ---------------------------------------------------------------------------
__device__ __forceinline__ void tdm_load_tile_2d(
    unsigned long long ga, unsigned int ldsOff, unsigned int w0,
    unsigned int tensor_d0, unsigned int tensor_d1,
    unsigned int tile_d0, unsigned int tile_d1, unsigned int stride0) {
    v4u g0;
    g0[0] = 1u;                                            // count=1 (valid D#)
    g0[1] = ldsOff;                                        // lds_addr [63:32]
    g0[2] = (unsigned int)ga;                              // global_addr [95:64]
    g0[3] = ((unsigned int)(ga >> 32) & 0x01FFFFFFu)       // global_addr [120:96]
            | (2u << 30);                                  // type=2 (image)
    v8i g1;
    g1[0] = (int)w0;                                       // mask/ds/pad flags
    g1[1] = (int)(tensor_d0 << 16);                        // tensor_dim0 lo16
    g1[2] = (int)(tensor_d1 << 16);                        // tensor_dim1 lo16
    g1[3] = (int)(tile_d0 << 16);                          // tile_dim0
    g1[4] = (int)tile_d1;                                  // tile_dim1 (tile_dim2=0)
    g1[5] = (int)stride0;                                  // tensor_dim0_stride lo32
    g1[6] = 0;
    g1[7] = 0;
    v4i z4 = (v4i){0, 0, 0, 0};
#if defined(__clang_major__) && (__clang_major__ >= 23)
    v8i z8 = (v8i){0, 0, 0, 0, 0, 0, 0, 0};
    __builtin_amdgcn_tensor_load_to_lds(g0, g1, z4, z4, z8, 0);
#else
    __builtin_amdgcn_tensor_load_to_lds(g0, g1, z4, z4, 0);
#endif
}

// ---------------------------------------------------------------------------
// Kernel 0: convert inputs to bf16; weights to bf16 TRANSPOSED (Wt[n][k])
// ---------------------------------------------------------------------------
__global__ void k0_convert(const float* __restrict__ X,
                           const float* __restrict__ Wq,
                           const float* __restrict__ Wk,
                           const float* __restrict__ Wv,
                           unsigned short* __restrict__ Xb,
                           unsigned short* __restrict__ Wt) {
    int tid = blockIdx.x * blockDim.x + threadIdx.x;
    if (tid < X_ELEMS) Xb[tid] = f2bf(X[tid]);
    if (tid < W_ELEMS) {
        int k = tid >> 9;          // row of W (contraction dim)
        int n = tid & 511;         // col of W
        int t = n * N_HIDDEN + k;  // transposed
        Wt[0 * W_ELEMS + t] = f2bf(Wq[tid]);
        Wt[1 * W_ELEMS + t] = f2bf(Wk[tid]);
        Wt[2 * W_ELEMS + t] = f2bf(Wv[tid]);
    }
}

// ---------------------------------------------------------------------------
// Kernel 1: QKV projection  out = relu(X @ W + b), bf16 WMMA, f32 accum.
//   grid = (ROWS/128, 512/64, 3), block = 256 (8 waves).
//   z==0 -> Q [B,H,S,D]; z==1 -> K [B,H,S,D]; z==2 -> V transposed [B,H,D,S]
// ---------------------------------------------------------------------------
__global__ void k1_qkv(const unsigned short* __restrict__ Xb,
                       const unsigned short* __restrict__ Wt3,
                       const float* __restrict__ bq,
                       const float* __restrict__ bk,
                       const float* __restrict__ bv,
                       unsigned short* __restrict__ Qb,
                       unsigned short* __restrict__ Kb,
                       unsigned short* __restrict__ Vt) {
    const int wave = threadIdx.x >> 5;
    const int lane = threadIdx.x & 31;
    const int col  = lane & 15;
    const int half = lane >> 4;

    const int z  = blockIdx.z;
    const int m0 = blockIdx.x * 128 + wave * 16;
    const int n0 = blockIdx.y * 64;

    const unsigned short* Wt = Wt3 + (size_t)z * W_ELEMS;
    const float* bias = (z == 0) ? bq : (z == 1) ? bk : bv;

    v8f acc[4];
#pragma unroll
    for (int t = 0; t < 4; ++t) acc[t] = (v8f){0.f,0.f,0.f,0.f,0.f,0.f,0.f,0.f};

    const int arow = m0 + col;
    for (int k0 = 0; k0 < N_HIDDEN; k0 += 32) {
        FragBF a;
#pragma unroll
        for (int v = 0; v < 8; ++v) {
            int k = k0 + kA(v, half);
            a.u[v] = *(const unsigned int*)(Xb + (size_t)arow * N_HIDDEN + k);
        }
#pragma unroll
        for (int t = 0; t < 4; ++t) {
            FragBF bm;
            int n = n0 + t * 16 + col;
#pragma unroll
            for (int v = 0; v < 8; ++v) {
                int k = k0 + kA(v, half);
                bm.u[v] = *(const unsigned int*)(Wt + (size_t)n * N_HIDDEN + k);
            }
            acc[t] = __builtin_amdgcn_wmma_f32_16x16x32_bf16(
                false, a.v, false, bm.v, (short)0, acc[t], false, false);
        }
    }

    // epilogue: +bias, ReLU, bf16, scatter to head layouts
#pragma unroll
    for (int t = 0; t < 4; ++t) {
        int n = n0 + t * 16 + col;
        int h = n >> 5;
        int d = n & 31;
        float bv_ = bias[n];
#pragma unroll
        for (int r = 0; r < 8; ++r) {
            int mg = m0 + r + 8 * half;
            float val = acc[t][r] + bv_;
            val = val > 0.f ? val : 0.f;
            unsigned short o = f2bf(val);
            int bi = mg >> 10;        // batch
            int s  = mg & 1023;       // seq
            if (z == 0)      Qb[(((size_t)(bi * HEADS + h)) * SEQ + s) * HDIM + d] = o;
            else if (z == 1) Kb[(((size_t)(bi * HEADS + h)) * SEQ + s) * HDIM + d] = o;
            else             Vt[(((size_t)(bi * HEADS + h)) * HDIM + d) * SEQ + s] = o;
        }
    }
}

// ---------------------------------------------------------------------------
// Kernel 2: flash attention per (b,h). grid=(SEQ/128, HEADS, BATCH), block=256.
//   K/V tiles staged to LDS via TDM (tensor_load_to_lds), double-buffered,
//   tracked with TENSORcnt.  Each wave owns 16 query rows; online softmax.
//   Writes y = attn_out + residual(x) to workspace (f32).
// ---------------------------------------------------------------------------
#define PSTRIDE 72   // bf16 row stride for P scratch
#define KSTR    34   // 32 elems + 1 DWORD TDM pad  (17 banks -> conflict-free)
#define VSTR    66   // 64 elems + 1 DWORD TDM pad  (33 banks -> conflict-free)

__global__ void k2_attn(const unsigned short* __restrict__ Qb,
                        const unsigned short* __restrict__ Kb,
                        const unsigned short* __restrict__ Vt,
                        const float* __restrict__ X,
                        float* __restrict__ Y) {
    __shared__ unsigned short ldsK[2][64 * KSTR];
    __shared__ unsigned short ldsV[2][32 * VSTR];
    __shared__ unsigned short ldsP[8 * 16 * PSTRIDE];

    const int wave = threadIdx.x >> 5;
    const int lane = threadIdx.x & 31;
    const int col  = lane & 15;
    const int half = lane >> 4;

    const int q0 = blockIdx.x * 128 + wave * 16;
    const int h  = blockIdx.y;
    const int b  = blockIdx.z;

    const size_t bh = (size_t)(b * HEADS + h);
    const unsigned short* Q = Qb + bh * SEQ * HDIM;
    const unsigned short* K = Kb + bh * SEQ * HDIM;
    const unsigned short* V = Vt + bh * HDIM * SEQ;  // [d][s]

    // TDM flag words: data_size=2B, pad_enable; pad_interval = row bytes
    const unsigned int w0K = (1u << 16) | (1u << 20) | (3u << 22);  // 64B rows
    const unsigned int w0V = (1u << 16) | (1u << 20) | (4u << 22);  // 128B rows

    // Q A-fragment: loop invariant (contraction over d == 32 exactly)
    FragBF qa;
    {
        int row = q0 + col;
#pragma unroll
        for (int v = 0; v < 8; ++v)
            qa.u[v] = *(const unsigned int*)(Q + (size_t)row * HDIM + kA(v, half));
    }

    v8f o0 = (v8f){0.f,0.f,0.f,0.f,0.f,0.f,0.f,0.f};
    v8f o1 = (v8f){0.f,0.f,0.f,0.f,0.f,0.f,0.f,0.f};
    float mrow[8], lrow[8];
#pragma unroll
    for (int r = 0; r < 8; ++r) { mrow[r] = -3.0e38f; lrow[r] = 0.f; }

    unsigned short* P = ldsP + wave * 16 * PSTRIDE;
    const float scale = 0.17677669529663687f;  // 1/sqrt(32)

    // ---- prologue: DMA first K/V tiles into buffer 0 ----
    if (wave == 0) {
        tdm_load_tile_2d((unsigned long long)(const void*)K,
                         (unsigned int)(size_t)&ldsK[0][0], w0K,
                         HDIM, SEQ, HDIM, 64, HDIM);
        tdm_load_tile_2d((unsigned long long)(const void*)V,
                         (unsigned int)(size_t)&ldsV[0][0], w0V,
                         SEQ, HDIM, 64, HDIM, SEQ);
    }

    int buf = 0;
    for (int kb = 0; kb < SEQ; kb += 64) {
        const int nbuf = buf ^ 1;
        if (wave == 0) {
            if (kb + 64 < SEQ) {
                // prefetch next tiles into the other buffer, then wait for
                // the current pair only (tensor ops complete in order)
                tdm_load_tile_2d((unsigned long long)(const void*)(K + (size_t)(kb + 64) * HDIM),
                                 (unsigned int)(size_t)&ldsK[nbuf][0], w0K,
                                 HDIM, SEQ, HDIM, 64, HDIM);
                tdm_load_tile_2d((unsigned long long)(const void*)(V + (kb + 64)),
                                 (unsigned int)(size_t)&ldsV[nbuf][0], w0V,
                                 SEQ, HDIM, 64, HDIM, SEQ);
                __builtin_amdgcn_s_wait_tensorcnt(2);
            } else {
                __builtin_amdgcn_s_wait_tensorcnt(0);
            }
        }
        __syncthreads();  // current tiles resident in LDS for all waves

        const unsigned short* Kt = &ldsK[buf][0];
        const unsigned short* Vl = &ldsV[buf][0];

        // ---- scores: 4 tiles of 16 keys, one wmma each (K-dim = d = 32) ----
        v8f s[4];
#pragma unroll
        for (int t = 0; t < 4; ++t) {
            FragBF kf;
            int lkey = t * 16 + col;
#pragma unroll
            for (int v = 0; v < 8; ++v) {
                int d = kA(v, half);
                kf.u[v] = *(const unsigned int*)(Kt + lkey * KSTR + d);
            }
            v8f z = (v8f){0.f,0.f,0.f,0.f,0.f,0.f,0.f,0.f};
            s[t] = __builtin_amdgcn_wmma_f32_16x16x32_bf16(
                false, qa.v, false, kf.v, (short)0, z, false, false);
        }

        // ---- online softmax over this 64-key block ----
        float lm[8];
#pragma unroll
        for (int r = 0; r < 8; ++r) {
            float m0v = s[0][r] * scale;
#pragma unroll
            for (int t = 1; t < 4; ++t) {
                float sv = s[t][r] * scale;
                m0v = sv > m0v ? sv : m0v;
            }
            lm[r] = m0v;
        }
#pragma unroll
        for (int r = 0; r < 8; ++r) {
#pragma unroll
            for (int msk = 1; msk <= 8; msk <<= 1) {
                float o = __shfl_xor(lm[r], msk, 32);
                lm[r] = o > lm[r] ? o : lm[r];
            }
        }
        float corr[8], ls[8];
#pragma unroll
        for (int r = 0; r < 8; ++r) {
            float nm = mrow[r] > lm[r] ? mrow[r] : lm[r];
            corr[r] = __expf(mrow[r] - nm);
            mrow[r] = nm;
            ls[r] = 0.f;
        }
#pragma unroll
        for (int t = 0; t < 4; ++t)
#pragma unroll
            for (int r = 0; r < 8; ++r) {
                float p = __expf(s[t][r] * scale - mrow[r]);
                s[t][r] = p;
                ls[r] += p;
            }
#pragma unroll
        for (int r = 0; r < 8; ++r) {
#pragma unroll
            for (int msk = 1; msk <= 8; msk <<= 1)
                ls[r] += __shfl_xor(ls[r], msk, 32);
            lrow[r] = lrow[r] * corr[r] + ls[r];
        }
#pragma unroll
        for (int r = 0; r < 8; ++r) { o0[r] *= corr[r]; o1[r] *= corr[r]; }

        // ---- relayout P: C-layout regs -> LDS -> A-layout fragments ----
#pragma unroll
        for (int t = 0; t < 4; ++t)
#pragma unroll
            for (int r = 0; r < 8; ++r)
                P[(r + 8 * half) * PSTRIDE + t * 16 + col] = f2bf(s[t][r]);
        asm volatile("s_wait_dscnt 0" ::: "memory");

        FragBF pa[2];
#pragma unroll
        for (int f = 0; f < 2; ++f)
#pragma unroll
            for (int v = 0; v < 8; ++v) {
                int k = f * 32 + kA(v, half);
                pa[f].u[v] = *(const unsigned int*)(P + col * PSTRIDE + k);
            }

        // ---- O += P @ V  (2 key-frags x 2 d-tiles) ----
#pragma unroll
        for (int f = 0; f < 2; ++f) {
            FragBF vb0, vb1;
#pragma unroll
            for (int v = 0; v < 8; ++v) {
                int lkey = f * 32 + kA(v, half);
                vb0.u[v] = *(const unsigned int*)(Vl + col * VSTR + lkey);
                vb1.u[v] = *(const unsigned int*)(Vl + (16 + col) * VSTR + lkey);
            }
            o0 = __builtin_amdgcn_wmma_f32_16x16x32_bf16(
                false, pa[f].v, false, vb0.v, (short)0, o0, false, false);
            o1 = __builtin_amdgcn_wmma_f32_16x16x32_bf16(
                false, pa[f].v, false, vb1.v, (short)0, o1, false, false);
        }
        __syncthreads();  // all waves done with buf before it is re-filled
        buf = nbuf;
    }

    // ---- epilogue: normalize, add residual, write y ----
#pragma unroll
    for (int r = 0; r < 8; ++r) {
        float inv = 1.0f / lrow[r];
        int qr = q0 + r + 8 * half;
        size_t base = ((size_t)b * SEQ + qr) * N_HIDDEN + h * HDIM;
        float y0 = o0[r] * inv + X[base + col];
        float y1 = o1[r] * inv + X[base + 16 + col];
        Y[base + col]      = y0;
        Y[base + 16 + col] = y1;
    }
}

// ---------------------------------------------------------------------------
// Kernel 3: BatchNorm stats per channel (mean, rstd) over 8192 rows
// ---------------------------------------------------------------------------
__global__ void k3_bnstats(const float* __restrict__ Y,
                           float* __restrict__ Mean,
                           float* __restrict__ Rstd) {
    __shared__ float s1[256], s2[256];
    const int c = blockIdx.x;
    float sum = 0.f, sq = 0.f;
    for (int i = threadIdx.x; i < ROWS; i += 256) {
        float v = Y[(size_t)i * N_HIDDEN + c];
        sum += v;
        sq  += v * v;
    }
    s1[threadIdx.x] = sum;
    s2[threadIdx.x] = sq;
    __syncthreads();
    for (int off = 128; off > 0; off >>= 1) {
        if (threadIdx.x < off) {
            s1[threadIdx.x] += s1[threadIdx.x + off];
            s2[threadIdx.x] += s2[threadIdx.x + off];
        }
        __syncthreads();
    }
    if (threadIdx.x == 0) {
        float mean = s1[0] * (1.0f / ROWS);
        float var  = s2[0] * (1.0f / ROWS) - mean * mean;
        Mean[c] = mean;
        Rstd[c] = rsqrtf(var + 1e-3f);
    }
}

// ---------------------------------------------------------------------------
// Kernel 4: apply BN affine, float4 vectorized
// ---------------------------------------------------------------------------
__global__ void k4_bnapply(const float* __restrict__ Y,
                           const float* __restrict__ Mean,
                           const float* __restrict__ Rstd,
                           const float* __restrict__ Gamma,
                           const float* __restrict__ Beta,
                           float* __restrict__ Out) {
    int i4 = blockIdx.x * blockDim.x + threadIdx.x;
    if (i4 >= X_ELEMS / 4) return;
    int e  = i4 * 4;
    int c0 = e & 511;
    float4 y = *(const float4*)(Y + e);
    float4 m = *(const float4*)(Mean + c0);
    float4 r = *(const float4*)(Rstd + c0);
    float4 g = *(const float4*)(Gamma + c0);
    float4 bt = *(const float4*)(Beta + c0);
    float4 o;
    o.x = (y.x - m.x) * r.x * g.x + bt.x;
    o.y = (y.y - m.y) * r.y * g.y + bt.y;
    o.z = (y.z - m.z) * r.z * g.z + bt.z;
    o.w = (y.w - m.w) * r.w * g.w + bt.w;
    *(float4*)(Out + e) = o;
}

// ---------------------------------------------------------------------------
// Host launcher
// ---------------------------------------------------------------------------
extern "C" void kernel_launch(void* const* d_in, const int* in_sizes, int n_in,
                              void* d_out, int out_size, void* d_ws, size_t ws_size,
                              hipStream_t stream) {
    const float* X     = (const float*)d_in[0];
    const float* Wq    = (const float*)d_in[1];
    const float* bq    = (const float*)d_in[2];
    const float* Wk    = (const float*)d_in[3];
    const float* bk    = (const float*)d_in[4];
    const float* Wv    = (const float*)d_in[5];
    const float* bv    = (const float*)d_in[6];
    const float* gamma = (const float*)d_in[7];
    const float* beta  = (const float*)d_in[8];
    float* Out = (float*)d_out;

    char* w = (char*)d_ws;
    unsigned short* Xb  = (unsigned short*)(w);                        //  8.0 MB
    unsigned short* Wt  = (unsigned short*)(w + 8388608);              //  1.5 MB (x3, transposed)
    unsigned short* Qb  = (unsigned short*)(w + 9961472);              //  8.0 MB [B,H,S,D]
    unsigned short* Kb  = (unsigned short*)(w + 18350080);             //  8.0 MB [B,H,S,D]
    unsigned short* Vt  = (unsigned short*)(w + 26738688);             //  8.0 MB [B,H,D,S]
    float*          Y   = (float*)(w + 35127296);                      // 16.0 MB
    float*          Mn  = (float*)(w + 51904512);
    float*          Rs  = (float*)(w + 51906560);

    k0_convert<<<X_ELEMS / 256, 256, 0, stream>>>(X, Wq, Wk, Wv, Xb, Wt);
    k1_qkv<<<dim3(ROWS / 128, N_HIDDEN / 64, 3), 256, 0, stream>>>(
        Xb, Wt, bq, bk, bv, Qb, Kb, Vt);
    k2_attn<<<dim3(SEQ / 128, HEADS, BATCH), 256, 0, stream>>>(Qb, Kb, Vt, X, Y);
    k3_bnstats<<<N_HIDDEN, 256, 0, stream>>>(Y, Mn, Rs);
    k4_bnapply<<<(X_ELEMS / 4 + 255) / 256, 256, 0, stream>>>(Y, Mn, Rs, gamma, beta, Out);
}
